// GINDropEncoder_38319698215465
// MI455X (gfx1250) — matrix operand order repaired and checked by reading, not compiled, not run on hardware
//
#include <hip/hip_runtime.h>
#include <hip/hip_bf16.h>
#include <math.h>

// ---------------- problem constants ----------------
#define N_NODES 50000
#define E_EDGES 800000
#define DF      64
#define RREP    4
#define NTILES  (N_NODES / 16)   // 3125, exact
#define P_DROP_U32 858993459u    // 0.2 * 2^32

typedef __attribute__((ext_vector_type(16))) _Float16 v16h;
typedef __attribute__((ext_vector_type(8)))  float    v8f;

// ---------------- helpers ----------------
// mish(x) = x * tanh(softplus(x)).  With t = 1 + e^x:
// tanh(ln t) = (t^2 - 1) / (t^2 + 1)  (exact), so one exp + one rcp suffices.
__device__ __forceinline__ float mish_f(float x) {
  if (x > 20.0f) return x;                 // tanh(softplus) == 1 to fp32
  float t  = 1.0f + __expf(x);             // x <= 20 -> no overflow
  float t2 = t * t;
  return x * (t2 - 1.0f) * __builtin_amdgcn_rcpf(t2 + 1.0f);
}

// deterministic per-(layer, replica, node) keep mask; keep with prob 0.8
__device__ __forceinline__ float keep_mask(int layer, int r, unsigned n) {
  unsigned x = n * 0x9E3779B1u ^ ((unsigned)layer * 0x85EBCA77u)
             ^ ((unsigned)r * 0xC2B2AE3Du) ^ 0x27220A95u;
  x ^= x >> 15; x *= 0x2C1B3C6Du;
  x ^= x >> 12; x *= 0x297A2D39u;
  x ^= x >> 15;
  return (x >= P_DROP_U32) ? 1.0f : 0.0f;
}

// A fragment (16x32 f16) from row-major f32, stride DF.
// ISA layout: lanes 0-15 halves 0-7 = K k0+0..7, halves 8-15 = K k0+16..23;
//             lanes 16-31 shifted by +8.
__device__ __forceinline__ v16h make_A_f32(const float* __restrict__ base,
                                           int lane, int k0) {
  int row = lane & 15;
  int kb  = k0 + ((lane >> 4) << 3);
  const float* p = base + row * DF + kb;
  v16h a;
#pragma unroll
  for (int j = 0; j < 8; ++j) {
    a[j]     = (_Float16)p[j];
    a[j + 8] = (_Float16)p[j + 16];
  }
  return a;
}

// A fragment for hg = h*keep + agg (per-lane row already resolved)
__device__ __forceinline__ v16h make_A_hg(const float* __restrict__ hrow,
                                          const float* __restrict__ aggrow,
                                          float km, int lane, int k0) {
  int kb = k0 + ((lane >> 4) << 3);
  v16h a;
#pragma unroll
  for (int j = 0; j < 8; ++j) {
    a[j]     = (_Float16)(hrow[kb + j]      * km + aggrow[kb + j]);
    a[j + 8] = (_Float16)(hrow[kb + 16 + j] * km + aggrow[kb + 16 + j]);
  }
  return a;
}

// 16x64 = (16x64)@(64x64) + bias. Weights pre-swizzled into fragment order:
// wm[frag(=t*2+kstep)][lane] is one contiguous v16h (32B) per lane.
__device__ __forceinline__ void gemm_tile(v16h a0, v16h a1,
                                          const _Float16* __restrict__ wm,
                                          const float* __restrict__ bias,
                                          int lane, v8f* acc) {
  const v16h* wb = (const v16h*)wm;
#pragma unroll
  for (int t = 0; t < 4; ++t) {
    float bv = bias[t * 16 + (lane & 15)];
    v8f c;
#pragma unroll
    for (int j = 0; j < 8; ++j) c[j] = bv;
    v16h b0 = wb[(t * 2 + 0) * 32 + lane];
    c = __builtin_amdgcn_wmma_f32_16x16x32_f16(false, a0, false, b0,
                                               (short)0, c, false, false);
    v16h b1 = wb[(t * 2 + 1) * 32 + lane];
    c = __builtin_amdgcn_wmma_f32_16x16x32_f16(false, a1, false, b1,
                                               (short)0, c, false, false);
    acc[t] = c;
  }
}

// mish + store C (16x64) into LDS directly in A-fragment order:
// element (M,K): kstep=K>>5, kk=K&31, owner lane = M + 16*((kk>>3)&1),
// half j = (kk&7) + 8*(kk>>4).  s layout: [kstep][lane][16 halves].
__device__ __forceinline__ void stage_mish_frag(_Float16* __restrict__ s,
                                                const v8f* acc, int lane) {
  int mbase = (lane >> 4) << 3;
  int colb  = lane & 15;
#pragma unroll
  for (int t = 0; t < 4; ++t)
#pragma unroll
    for (int rr = 0; rr < 8; ++rr) {
      int M = mbase + rr;
      int K = t * 16 + colb;
      int kstep = K >> 5;
      int kk = K & 31;
      int laneSel = (kk >> 3) & 1;
      int j = (kk & 7) + ((kk >> 4) << 3);
      s[kstep * 512 + (M + 16 * laneSel) * 16 + j] = (_Float16)mish_f(acc[t][rr]);
    }
}

// ---------------- kernels ----------------
// Pre-swizzle all 8 weight matrices into per-lane WMMA B fragments.
// wfrag layout: [matrix(8)][frag(8)=t*2+kstep][lane(32)][16 halves]
__global__ void prep_weights_kernel(const float* __restrict__ nW1,
                                    const float* __restrict__ nW2,
                                    const float* __restrict__ cW1,
                                    const float* __restrict__ cW2,
                                    _Float16* __restrict__ wfrag) {
  int i = blockIdx.x * blockDim.x + threadIdx.x;  // (m, f, lane)
  if (i >= 8 * 8 * 32) return;
  int m    = i >> 8;
  int f    = (i >> 5) & 7;
  int lane = i & 31;
  int t = f >> 1, kstep = f & 1;
  const float* w;
  if      (m == 0) w = nW1;
  else if (m == 1) w = nW2;
  else if (m <  5) w = cW1 + (size_t)(m - 2) * 4096;
  else             w = cW2 + (size_t)(m - 5) * 4096;
  int col = t * 16 + (lane & 15);
  int kb  = kstep * 32 + ((lane >> 4) << 3);   // lanes 0-15: K 0..7,16..23; 16-31: K 8..15,24..31
  _Float16* out = wfrag + (size_t)m * 4096 + f * 512 + lane * 16;
#pragma unroll
  for (int j = 0; j < 16; ++j) out[j] = (_Float16)w[(kb + (j & 7) + ((j >> 3) << 4)) * DF + col];
}

__global__ void zero_kernel(float* __restrict__ p, int count) {
  int i = blockIdx.x * blockDim.x + threadIdx.x;
  if (i < count) p[i] = 0.0f;
}

// x_proj = mish(x@W1+b1)@W2+b2 ; write to d_out (x_proj) and h (init)
__global__ void __launch_bounds__(256)
node_mlp_kernel(const float* __restrict__ x,
                const _Float16* __restrict__ wfrag,
                const float* __restrict__ b1, const float* __restrict__ b2,
                float* __restrict__ xproj, float* __restrict__ h) {
  __shared__ v16h stage[8][64];   // per wave: 2 fragments x 32 lanes
  int lane = threadIdx.x & 31;
  int wid  = threadIdx.x >> 5;
  int tile = blockIdx.x * 8 + wid;
  if (tile >= NTILES) return;
  _Float16* s = (_Float16*)stage[wid];
  const v16h* sv = (const v16h*)s;
  const float* xt = x + (size_t)tile * 16 * DF;

  v16h a0 = make_A_f32(xt, lane, 0);
  v16h a1 = make_A_f32(xt, lane, 32);
  v8f acc[4];
  gemm_tile(a0, a1, wfrag, b1, lane, acc);              // MLP1
  stage_mish_frag(s, acc, lane);                        // mish -> LDS (frag order)
  a0 = sv[lane];
  a1 = sv[32 + lane];
  gemm_tile(a0, a1, wfrag + 4096, b2, lane, acc);       // MLP2

  int mbase = (lane >> 4) << 3, col = lane & 15;
#pragma unroll
  for (int t = 0; t < 4; ++t)
#pragma unroll
    for (int rr = 0; rr < 8; ++rr) {
      size_t idx = ((size_t)tile * 16 + mbase + rr) * DF + t * 16 + col;
      float v = acc[t][rr];
      xproj[idx] = v;
      h[idx]     = v;
    }
}

// edge scatter: agg[r, dst, :] += h[src, :] * keep(layer, r, src)
__global__ void scatter_kernel(const float* __restrict__ h,
                               const int* __restrict__ edges,
                               float* __restrict__ agg, int layer) {
  long long gid = (long long)blockIdx.x * blockDim.x + threadIdx.x;
  if (gid >= (long long)E_EDGES * DF) return;
  int e = (int)(gid >> 6);
  int d = (int)(gid & 63);
  int src = edges[e];
  int dst = edges[E_EDGES + e];
  float hv = h[(size_t)src * DF + d];
#pragma unroll
  for (int r = 0; r < RREP; ++r) {
    if (keep_mask(layer, r, (unsigned)src) != 0.0f)
      unsafeAtomicAdd(&agg[((size_t)r * N_NODES + dst) * DF + d], hv);
  }
}

// per tile: loop replicas, hg = h*keep + agg, 2 fused GEMMs, mean, +h
__global__ void __launch_bounds__(256)
conv_layer_kernel(const float* __restrict__ h,
                  const float* __restrict__ agg,
                  const _Float16* __restrict__ w1,
                  const _Float16* __restrict__ w2,
                  const float* __restrict__ b1, const float* __restrict__ b2,
                  float* __restrict__ h_pre, int layer) {
  __shared__ v16h stage[8][64];
  int lane = threadIdx.x & 31;
  int wid  = threadIdx.x >> 5;
  int tile = blockIdx.x * 8 + wid;
  if (tile >= NTILES) return;
  _Float16* s = (_Float16*)stage[wid];
  const v16h* sv = (const v16h*)s;

  unsigned node_a = (unsigned)tile * 16 + (lane & 15);
  const float* hrow = h + (size_t)node_a * DF;

  v8f sum[4];
#pragma unroll
  for (int t = 0; t < 4; ++t)
#pragma unroll
    for (int j = 0; j < 8; ++j) sum[t][j] = 0.0f;

  for (int r = 0; r < RREP; ++r) {
    float km = keep_mask(layer, r, node_a);
    const float* aggrow = agg + ((size_t)r * N_NODES + node_a) * DF;
    v16h a0 = make_A_hg(hrow, aggrow, km, lane, 0);
    v16h a1 = make_A_hg(hrow, aggrow, km, lane, 32);
    v8f acc[4];
    gemm_tile(a0, a1, w1, b1, lane, acc);
    stage_mish_frag(s, acc, lane);
    a0 = sv[lane];
    a1 = sv[32 + lane];
    gemm_tile(a0, a1, w2, b2, lane, acc);
#pragma unroll
    for (int t = 0; t < 4; ++t) sum[t] += acc[t];
  }

  int mbase = (lane >> 4) << 3, col = lane & 15;
#pragma unroll
  for (int t = 0; t < 4; ++t)
#pragma unroll
    for (int rr = 0; rr < 8; ++rr) {
      size_t idx = ((size_t)tile * 16 + mbase + rr) * DF + t * 16 + col;
      h_pre[idx] = sum[t][rr] * 0.25f + h[idx];
    }
}

// per-feature sums for batchnorm: stats[0:64]=sum, stats[64:128]=sumsq
__global__ void __launch_bounds__(256)
bn_stats_kernel(const float* __restrict__ h_pre, float* __restrict__ stats) {
  __shared__ float sb[256], sb2[256];
  int tid  = threadIdx.x;
  int d    = tid & 63;
  int rsub = tid >> 6;  // 0..3
  float s = 0.0f, s2 = 0.0f;
  for (int row = blockIdx.x * 4 + rsub; row < N_NODES; row += gridDim.x * 4) {
    float v = h_pre[(size_t)row * DF + d];
    s += v; s2 += v * v;
  }
  sb[tid] = s; sb2[tid] = s2;
  __syncthreads();
  if (tid < 64) {
    float ts  = sb[tid]  + sb[tid + 64]  + sb[tid + 128]  + sb[tid + 192];
    float ts2 = sb2[tid] + sb2[tid + 64] + sb2[tid + 128] + sb2[tid + 192];
    unsafeAtomicAdd(&stats[d],      ts);
    unsafeAtomicAdd(&stats[64 + d], ts2);
  }
}

// h = mish((h_pre - mu) * rsqrt(var + eps)); last layer: d_out = x_proj + h
__global__ void bn_apply_kernel(const float* __restrict__ h_pre,
                                const float* __restrict__ stats,
                                float* __restrict__ h,
                                float* __restrict__ dout, int last) {
  long long idx = (long long)blockIdx.x * blockDim.x + threadIdx.x;
  if (idx >= (long long)N_NODES * DF) return;
  int d = (int)(idx & 63);
  const float invN = 1.0f / (float)N_NODES;
  float mu  = stats[d] * invN;
  float var = stats[64 + d] * invN - mu * mu;
  float v = (h_pre[idx] - mu) * __builtin_amdgcn_rsqf(var + 1e-5f);
  v = mish_f(v);
  if (last) dout[idx] = dout[idx] + v;
  else      h[idx]    = v;
}

// ---------------- launch ----------------
extern "C" void kernel_launch(void* const* d_in, const int* in_sizes, int n_in,
                              void* d_out, int out_size, void* d_ws, size_t ws_size,
                              hipStream_t stream) {
  const float* x   = (const float*)d_in[0];
  const int*   edg = (const int*)  d_in[1];
  const float* nW1 = (const float*)d_in[2];
  const float* nb1 = (const float*)d_in[3];
  const float* nW2 = (const float*)d_in[4];
  const float* nb2 = (const float*)d_in[5];
  const float* cW1 = (const float*)d_in[6];
  const float* cb1 = (const float*)d_in[7];
  const float* cW2 = (const float*)d_in[8];
  const float* cb2 = (const float*)d_in[9];
  float* dout = (float*)d_out;

  const size_t ND4  = (size_t)N_NODES * DF * sizeof(float);   // 12.8 MB
  const size_t RND4 = (size_t)RREP * N_NODES * DF * sizeof(float);
  char* ws = (char*)d_ws;
  _Float16* wfrag = (_Float16*)ws;                         // 64 KB, fragment order
  float*    h     = (float*)(ws + 65536);
  float*    h_pre = (float*)(ws + 65536 + ND4);
  float*    agg   = (float*)(ws + 65536 + 2 * ND4);        // R*N*DF
  float*    stats = (float*)(ws + 65536 + 2 * ND4 + RND4); // 128 floats (after agg)

  prep_weights_kernel<<<(8 * 8 * 32 + 255) / 256, 256, 0, stream>>>(nW1, nW2, cW1, cW2, wfrag);

  const int tileBlocks = (NTILES + 7) / 8;  // 8 waves/block, 1 tile/wave
  node_mlp_kernel<<<tileBlocks, 256, 0, stream>>>(x, wfrag, nb1, nb2, dout, h);

  const int zeroCount = RREP * N_NODES * DF + 128;  // agg + stats contiguous
  const long long scatterThreads = (long long)E_EDGES * DF;

  for (int i = 0; i < 3; ++i) {
    zero_kernel<<<(zeroCount + 255) / 256, 256, 0, stream>>>(agg, zeroCount);
    scatter_kernel<<<(unsigned)((scatterThreads + 255) / 256), 256, 0, stream>>>(h, edg, agg, i);
    conv_layer_kernel<<<tileBlocks, 256, 0, stream>>>(
        h, agg, wfrag + (size_t)(2 + i) * 4096, wfrag + (size_t)(5 + i) * 4096,
        cb1 + i * DF, cb2 + i * DF, h_pre, i);
    bn_stats_kernel<<<512, 256, 0, stream>>>(h_pre, stats);
    bn_apply_kernel<<<(N_NODES * DF + 255) / 256, 256, 0, stream>>>(
        h_pre, stats, h, dout, (i == 2) ? 1 : 0);
  }
}
